// GCNLinkPredictorAblation_64020782514259
// MI455X (gfx1250) — matrix operand order repaired and checked by reading, not compiled, not run on hardware
//
#include <hip/hip_runtime.h>

// ---------------------------------------------------------------------------
// Types for CDNA5 WMMA (wave32): v_wmma_f32_16x16x32_bf16
// ---------------------------------------------------------------------------
typedef __attribute__((ext_vector_type(16))) __bf16 v16bf;
typedef __attribute__((ext_vector_type(8)))  float  v8f;

union BF16Frag {
    v16bf v;
    unsigned short u[16];
    int4 q[2];
};

__device__ __forceinline__ unsigned short f2bf(float f) {
    // round-to-nearest-even fp32 -> bf16
    unsigned int u = __float_as_uint(f);
    u += 0x7FFFu + ((u >> 16) & 1u);
    return (unsigned short)(u >> 16);
}

__device__ __forceinline__ v8f zero8() {
    v8f z = {0.f, 0.f, 0.f, 0.f, 0.f, 0.f, 0.f, 0.f};
    return z;
}

#define BN_EPS 1e-5f

// ---------------------------------------------------------------------------
// Small elementwise / scatter kernels
// ---------------------------------------------------------------------------
__global__ void k_fill1(float* p, int n) {
    int i = blockIdx.x * blockDim.x + threadIdx.x;
    if (i < n) p[i] = 1.0f;  // self-loop contributes 1 to every degree
}

__global__ void k_deg_scatter(const int* __restrict__ dst, float* deg, int E) {
    int e = blockIdx.x * blockDim.x + threadIdx.x;
    if (e < E) atomicAdd(&deg[dst[e]], 1.0f);
}

__global__ void k_dinv(const float* __restrict__ deg, float* dinv, int n) {
    int i = blockIdx.x * blockDim.x + threadIdx.x;
    if (i < n) dinv[i] = rsqrtf(deg[i]);  // deg >= 1 always (self loop)
}

// W [K][Nc] f32 row-major  ->  Wt [Nc][K] bf16 row-major (i.e. transposed)
__global__ void k_cvt_transpose(const float* __restrict__ W, unsigned short* __restrict__ Wt,
                                int K, int Nc) {
    int idx = blockIdx.x * blockDim.x + threadIdx.x;
    if (idx >= K * Nc) return;
    int n = idx / K, k = idx % K;
    Wt[idx] = f2bf(W[(size_t)k * Nc + n]);
}

// AGG[i][f] = H[i][f] * dinv[i]^2   (self-loop term of symmetric-normalized A)
__global__ void k_agg_init(const float* __restrict__ H, const float* __restrict__ dinv,
                           float* __restrict__ AGG, int total) {
    int idx = blockIdx.x * blockDim.x + threadIdx.x;
    if (idx >= total) return;
    int i = idx >> 7;
    float d = dinv[i];
    AGG[idx] = H[idx] * d * d;
}

// AGG[dst][f] += H[src][f] * dinv[src]*dinv[dst]  over all edges
__global__ void k_agg_scatter(const float* __restrict__ H, const int* __restrict__ src,
                              const int* __restrict__ dst, const float* __restrict__ dinv,
                              float* AGG, int E) {
    int gid = blockIdx.x * blockDim.x + threadIdx.x;
    if (gid >= E * 128) return;
    int e = gid >> 7, f = gid & 127;
    int s = src[e], d = dst[e];
    float w = dinv[s] * dinv[d];
    atomicAdd(&AGG[(size_t)d * 128 + f], H[(size_t)s * 128 + f] * w);
}

// h1 = bf16( relu( (AGG + b) * gamma/sqrt(1+eps) + beta ) )
__global__ void k_finalize1(const float* __restrict__ AGG, const float* __restrict__ b,
                            const float* __restrict__ g, const float* __restrict__ beta,
                            unsigned short* __restrict__ h1bf, int total) {
    int idx = blockIdx.x * blockDim.x + threadIdx.x;
    if (idx >= total) return;
    int f = idx & 127;
    float scale = g[f] * rsqrtf(1.0f + BN_EPS);
    float v = (AGG[idx] + b[f]) * scale + beta[f];
    v = fmaxf(v, 0.0f);
    h1bf[idx] = f2bf(v);
}

// h = (AGG + b) * gamma/sqrt(1+eps) + beta   (no relu, f32 out)
__global__ void k_finalize2(const float* __restrict__ AGG, const float* __restrict__ b,
                            const float* __restrict__ g, const float* __restrict__ beta,
                            float* __restrict__ hout, int total) {
    int idx = blockIdx.x * blockDim.x + threadIdx.x;
    if (idx >= total) return;
    int f = idx & 127;
    float scale = g[f] * rsqrtf(1.0f + BN_EPS);
    hout[idx] = (AGG[idx] + b[f]) * scale + beta[f];
}

// ---------------------------------------------------------------------------
// WMMA GEMM: Out[M][128] = A[M][K] @ B[K][128]   (no bias)
// 256 threads = 8 waves per block, block covers a 64(M) x 128(N) tile.
// wave -> (m-subtile = wave>>1, n-half = wave&1); each wave: 16x64 quadrant
// with only 4 live accumulators (32 VGPRs) -> no spills.
// A_IS_F32: A is f32 (converted to bf16 in registers); else A is bf16.
// Bt is B transposed to [128][K] bf16 so B-fragments are contiguous.
//
// A-fragment layout (16-bit A 16x32): lane<16 row=lane, elems 0-7 = K+0..7,
// elems 8-15 = K+16..23; lane>=16 row=lane-16, K offset +8 / +24.
// B-fragment: lane n<16: elems 0-7 = B[K+0..7][n], 8-15 = B[K+16..23][n]; lane>=16: K offset +8 / +24.
// ---------------------------------------------------------------------------
template <bool A_IS_F32>
__global__ void __launch_bounds__(256) k_gemm_n128(const void* __restrict__ Aptr,
                                                   const unsigned short* __restrict__ Bt,
                                                   float* __restrict__ Out, int M, int K) {
    const int wave   = threadIdx.x >> 5;
    const int lane   = threadIdx.x & 31;
    const int lane15 = lane & 15;
    const int hi     = lane >> 4;
    const int m0     = blockIdx.x * 64 + (wave >> 1) * 16;
    const int nbase  = (wave & 1) * 64;
    int arow = m0 + lane15;
    if (arow > M - 1) arow = M - 1;

    v8f acc[4];
#pragma unroll
    for (int i = 0; i < 4; ++i) acc[i] = zero8();

    for (int k = 0; k < K; k += 32) {
        BF16Frag a;
        if (A_IS_F32) {
            const float* ar = (const float*)Aptr + (size_t)arow * K + k + hi * 8;
            float4 q0 = *(const float4*)(ar);
            float4 q1 = *(const float4*)(ar + 4);
            float4 q2 = *(const float4*)(ar + 16);
            float4 q3 = *(const float4*)(ar + 20);
            float t[16] = {q0.x, q0.y, q0.z, q0.w, q1.x, q1.y, q1.z, q1.w,
                           q2.x, q2.y, q2.z, q2.w, q3.x, q3.y, q3.z, q3.w};
#pragma unroll
            for (int i = 0; i < 16; ++i) a.u[i] = f2bf(t[i]);
        } else {
            const unsigned short* ar =
                (const unsigned short*)Aptr + (size_t)arow * K + k + hi * 8;
            a.q[0] = *(const int4*)(ar);
            a.q[1] = *(const int4*)(ar + 16);
        }
#pragma unroll
        for (int nt = 0; nt < 4; ++nt) {
            const int col = nbase + nt * 16 + lane15;
            const unsigned short* br = Bt + (size_t)col * K + k + hi * 8;
            BF16Frag b;
            b.q[0] = *(const int4*)(br);
            b.q[1] = *(const int4*)(br + 16);
            acc[nt] = __builtin_amdgcn_wmma_f32_16x16x32_bf16(
                false, a.v, false, b.v, (short)0, acc[nt], false, false);
        }
    }
    // C/D layout: VGPR r -> row m0 + hi*8 + r, col = nbase + nt*16 + lane15
#pragma unroll
    for (int nt = 0; nt < 4; ++nt) {
        const int col = nbase + nt * 16 + lane15;
#pragma unroll
        for (int r = 0; r < 8; ++r) {
            const int m = m0 + hi * 8 + r;
            if (m < M) Out[(size_t)m * 128 + col] = acc[nt][r];
        }
    }
}

// ---------------------------------------------------------------------------
// Fused link decoder, one wave per 16-pair tile:
//   feat = [|h_u - h_v|, h_u * h_v]  staged in LDS (16x256 bf16, 8KB/wave)
//   z1 = relu(feat @ mW1 + mb1)      n-tile-outer WMMA (1 live accumulator)
//   z2 = relu(z1 @ mW2 + mb2)        n-tile-outer WMMA against z1 LDS tile
//   logit = z2 @ mW3 + mb3           VALU dot from LDS
// 4 waves per block, disjoint LDS slices -> no barriers needed (LDS ops are
// in-order within a wave). Per-wave LDS slice: feat 8KB + z1 4KB = 12KB;
// z2 (16x64 f32, 4KB) overlays the feat area once feat is consumed.
// ---------------------------------------------------------------------------
__device__ __forceinline__ int4 feat8v(const float* hu, const float* hv, int c0) {
    const int base = c0 & 127;  // runs of 8 never straddle the 128 boundary
    float4 u0 = *(const float4*)(hu + base);
    float4 u1 = *(const float4*)(hu + base + 4);
    float4 v0 = *(const float4*)(hv + base);
    float4 v1 = *(const float4*)(hv + base + 4);
    float fu[8] = {u0.x, u0.y, u0.z, u0.w, u1.x, u1.y, u1.z, u1.w};
    float fv[8] = {v0.x, v0.y, v0.z, v0.w, v1.x, v1.y, v1.z, v1.w};
    const bool prod = (c0 >= 128);
    union { int4 q; unsigned short u[8]; } r;
#pragma unroll
    for (int i = 0; i < 8; ++i) {
        float f = prod ? (fu[i] * fv[i]) : fabsf(fu[i] - fv[i]);
        r.u[i] = f2bf(f);
    }
    return r.q;
}

#define DEC_SLICE (16 * 256 + 16 * 128)  // ushorts per wave slice

__global__ void __launch_bounds__(128) k_decode(
        const float* __restrict__ h, const int* __restrict__ pairs, int P,
        const unsigned short* __restrict__ mW1t,  // [128][256] bf16
        const float* __restrict__ mb1,
        const unsigned short* __restrict__ mW2t,  // [64][128] bf16
        const float* __restrict__ mb2,
        const float* __restrict__ mW3,  // [64] f32
        const float* __restrict__ mb3,
        float* __restrict__ outp) {
    __shared__ __align__(16) unsigned short lds[4 * DEC_SLICE];  // 48KB

    const int wave    = threadIdx.x >> 5;
    const int lane    = threadIdx.x & 31;
    const int lane15  = lane & 15;
    const int hi      = lane >> 4;
    const int tile    = blockIdx.x * 4 + wave;
    const int rowBase = tile * 16;

    unsigned short* featp = lds + wave * DEC_SLICE;        // [16][256] bf16
    unsigned short* z1p   = featp + 16 * 256;              // [16][128] bf16
    float*          z2p   = (float*)featp;                 // [16][64] f32 (overlay)

    int pr = rowBase + lane15;
    if (pr > P - 1) pr = P - 1;
    const int u  = pairs[pr];
    const int vv = pairs[P + pr];
    const float* hu = h + (size_t)u * 128;
    const float* hv = h + (size_t)vv * 128;

    // ---- stage feat tile: lane (row=lane15) fills cols [hi*128, hi*128+128) ----
#pragma unroll
    for (int j = 0; j < 16; ++j) {
        const int c = hi * 128 + j * 8;
        *(int4*)(featp + lane15 * 256 + c) = feat8v(hu, hv, c);
    }

    // ---- layer 1: z1 = relu(feat @ mW1 + mb1), n-tile outer ----
    for (int nt = 0; nt < 8; ++nt) {
        v8f acc = zero8();
        const int col = nt * 16 + lane15;
#pragma unroll
        for (int kk = 0; kk < 8; ++kk) {
            BF16Frag a, b;
            const unsigned short* ar = featp + lane15 * 256 + kk * 32 + hi * 8;
            a.q[0] = *(const int4*)(ar);
            a.q[1] = *(const int4*)(ar + 16);
            const unsigned short* br = mW1t + (size_t)col * 256 + kk * 32 + hi * 8;
            b.q[0] = *(const int4*)(br);
            b.q[1] = *(const int4*)(br + 16);
            acc = __builtin_amdgcn_wmma_f32_16x16x32_bf16(
                false, a.v, false, b.v, (short)0, acc, false, false);
        }
#pragma unroll
        for (int r = 0; r < 8; ++r) {
            const int m = hi * 8 + r;
            float val = acc[r] + mb1[col];
            z1p[m * 128 + col] = f2bf(fmaxf(val, 0.0f));
        }
    }

    // ---- layer 2: z2 = relu(z1 @ mW2 + mb2), n-tile outer ----
    for (int nt = 0; nt < 4; ++nt) {
        v8f acc = zero8();
        const int col = nt * 16 + lane15;
#pragma unroll
        for (int kk = 0; kk < 4; ++kk) {
            BF16Frag a, b;
            const unsigned short* ar = z1p + lane15 * 128 + kk * 32 + hi * 8;
            a.q[0] = *(const int4*)(ar);
            a.q[1] = *(const int4*)(ar + 16);
            const unsigned short* br = mW2t + (size_t)col * 128 + kk * 32 + hi * 8;
            b.q[0] = *(const int4*)(br);
            b.q[1] = *(const int4*)(br + 16);
            acc = __builtin_amdgcn_wmma_f32_16x16x32_bf16(
                false, a.v, false, b.v, (short)0, acc, false, false);
        }
        // z2 overlays feat area; all feat reads are already done (in-order LDS)
#pragma unroll
        for (int r = 0; r < 8; ++r) {
            const int m = hi * 8 + r;
            float val = acc[r] + mb2[col];
            z2p[m * 64 + col] = fmaxf(val, 0.0f);
        }
    }

    // ---- layer 3: logit = z2 @ mW3 + mb3 ----
    if (lane < 16) {
        const int row = lane;
        float s = mb3[0];
#pragma unroll 8
        for (int k = 0; k < 64; ++k) s += z2p[row * 64 + k] * mW3[k];
        const int orow = rowBase + row;
        if (orow < P) outp[orow] = s;
    }
}

// ---------------------------------------------------------------------------
// Host launch
// ---------------------------------------------------------------------------
extern "C" void kernel_launch(void* const* d_in, const int* in_sizes, int n_in,
                              void* d_out, int out_size, void* d_ws, size_t ws_size,
                              hipStream_t stream) {
    const float* x    = (const float*)d_in[0];
    const int*   mei  = (const int*)d_in[1];
    const int*   pos  = (const int*)d_in[2];
    const int*   neg  = (const int*)d_in[3];
    const float* W1   = (const float*)d_in[4];
    const float* b1   = (const float*)d_in[5];
    const float* bn1g = (const float*)d_in[6];
    const float* bn1b = (const float*)d_in[7];
    const float* W2   = (const float*)d_in[8];
    const float* b2   = (const float*)d_in[9];
    const float* bn2g = (const float*)d_in[10];
    const float* bn2b = (const float*)d_in[11];
    const float* mW1  = (const float*)d_in[12];
    const float* mb1  = (const float*)d_in[13];
    const float* mW2  = (const float*)d_in[14];
    const float* mb2  = (const float*)d_in[15];
    const float* mW3  = (const float*)d_in[16];
    const float* mb3  = (const float*)d_in[17];

    const int IN = 1024, HID = 128;
    const int Nn = in_sizes[0] / IN;   // 50000
    const int E  = in_sizes[1] / 2;    // 600000
    const int P  = in_sizes[2] / 2;    // 200000

    // workspace carve-up (256B aligned)
    char*  ws  = (char*)d_ws;
    size_t off = 0;
    auto take = [&](size_t bytes) -> char* {
        char* p = ws + off;
        off = (off + bytes + 255) & ~(size_t)255;
        return p;
    };
    float*          H    = (float*)take((size_t)Nn * HID * 4);
    float*          AGG  = (float*)take((size_t)Nn * HID * 4);
    unsigned short* h1bf = (unsigned short*)take((size_t)Nn * HID * 2);
    float*          hfin = (float*)take((size_t)Nn * HID * 4);
    float*          deg  = (float*)take((size_t)Nn * 4);
    float*          dinv = (float*)take((size_t)Nn * 4);
    unsigned short* W1t  = (unsigned short*)take((size_t)HID * IN * 2);
    unsigned short* W2t  = (unsigned short*)take((size_t)HID * HID * 2);
    unsigned short* mW1t = (unsigned short*)take((size_t)HID * 2 * HID * 2);
    unsigned short* mW2t = (unsigned short*)take((size_t)(HID / 2) * HID * 2);

    const int T = 256;
    const int totNF = Nn * HID;

    // degrees & symmetric normalization
    k_fill1<<<(Nn + T - 1) / T, T, 0, stream>>>(deg, Nn);
    k_deg_scatter<<<(E + T - 1) / T, T, 0, stream>>>(mei + E, deg, E);
    k_dinv<<<(Nn + T - 1) / T, T, 0, stream>>>(deg, dinv, Nn);

    // weight transposes -> bf16
    k_cvt_transpose<<<(IN * HID + T - 1) / T, T, 0, stream>>>(W1, W1t, IN, HID);
    k_cvt_transpose<<<(HID * HID + T - 1) / T, T, 0, stream>>>(W2, W2t, HID, HID);
    k_cvt_transpose<<<(2 * HID * HID + T - 1) / T, T, 0, stream>>>(mW1, mW1t, 2 * HID, HID);
    k_cvt_transpose<<<(HID * (HID / 2) + T - 1) / T, T, 0, stream>>>(mW2, mW2t, HID, HID / 2);

    // layer 1: GEMM -> aggregate -> BN+ReLU (bf16 out)
    k_gemm_n128<true><<<(Nn + 63) / 64, 256, 0, stream>>>(x, W1t, H, Nn, IN);
    k_agg_init<<<(totNF + T - 1) / T, T, 0, stream>>>(H, dinv, AGG, totNF);
    k_agg_scatter<<<(E * 128 + T - 1) / T, T, 0, stream>>>(H, mei, mei + E, dinv, AGG, E);
    k_finalize1<<<(totNF + T - 1) / T, T, 0, stream>>>(AGG, b1, bn1g, bn1b, h1bf, totNF);

    // layer 2: GEMM -> aggregate -> BN (f32 out)
    k_gemm_n128<false><<<(Nn + 63) / 64, 256, 0, stream>>>(h1bf, W2t, H, Nn, HID);
    k_agg_init<<<(totNF + T - 1) / T, T, 0, stream>>>(H, dinv, AGG, totNF);
    k_agg_scatter<<<(E * 128 + T - 1) / T, T, 0, stream>>>(H, mei, mei + E, dinv, AGG, E);
    k_finalize2<<<(totNF + T - 1) / T, T, 0, stream>>>(AGG, b2, bn2g, bn2b, hfin, totNF);

    // decode pos + neg
    const int tiles   = (P + 15) / 16;
    const int dblocks = (tiles + 3) / 4;
    k_decode<<<dblocks, 128, 0, stream>>>(hfin, pos, P, mW1t, mb1, mW2t, mb2, mW3, mb3,
                                          (float*)d_out);
    k_decode<<<dblocks, 128, 0, stream>>>(hfin, neg, P, mW1t, mb1, mW2t, mb2, mW3, mb3,
                                          (float*)d_out + P);
}
